// ScaledDotSelfAttention_70832600646361
// MI455X (gfx1250) — matrix operand order repaired and checked by reading, not compiled, not run on hardware
//
#include <hip/hip_runtime.h>
#include <hip/hip_bf16.h>

// ---------------------------------------------------------------------------
// Types for CDNA5 WMMA (wave32): bf16 operands, f32 accumulate.
// ---------------------------------------------------------------------------
typedef __attribute__((ext_vector_type(16))) __bf16 v16bf;
typedef __attribute__((ext_vector_type(8)))  __bf16 v8bf;
typedef __attribute__((ext_vector_type(8)))  float  v8f;

struct alignas(16) bf16x8 { __bf16 v[8]; };
struct alignas(8)  bf16x4 { __bf16 v[4]; };
union  pk2 { __bf16 b[2]; unsigned int u; };

#define NB_B 8
#define NB_S 1024
#define NB_D 1024
#define NB_H 16
#define NB_U 64

// ---------------------------------------------------------------------------
// Fragment loaders (k-contiguous order used here, not the ISA table order):
//  A (16x32): lane row = lane&15; elements = one contiguous 16-elem run at
//             k = 16*khalf + [0..15]  (same k order as B)
//  B (32x16): lane col = lane&15; elements = one contiguous 16-elem run at
//             k = 16*khalf + [0..15]  (requires k-contiguous storage)
// ---------------------------------------------------------------------------
__device__ __forceinline__ v16bf load_afrag(const __bf16* prow, int khalf) {
    v8bf lo = *(const v8bf*)(prow + 16 * khalf);
    v8bf hi = *(const v8bf*)(prow + 16 * khalf + 8);
    return __builtin_shufflevector(lo, hi, 0,1,2,3,4,5,6,7,8,9,10,11,12,13,14,15);
}
__device__ __forceinline__ v16bf load_bfrag(const __bf16* pcol_khalf) {
    v8bf lo = *(const v8bf*)(pcol_khalf);
    v8bf hi = *(const v8bf*)(pcol_khalf + 8);
    return __builtin_shufflevector(lo, hi, 0,1,2,3,4,5,6,7,8,9,10,11,12,13,14,15);
}

__device__ __forceinline__ float redmax16(float x) {
    #pragma unroll
    for (int m = 1; m < 16; m <<= 1) x = fmaxf(x, __shfl_xor(x, m, 32));
    return x;
}
__device__ __forceinline__ float redsum16(float x) {
    #pragma unroll
    for (int m = 1; m < 16; m <<= 1) x += __shfl_xor(x, m, 32);
    return x;
}

// ---------------------------------------------------------------------------
// Kernel 1: x = inputs * mask, cast fp32 -> bf16   [B,S,D]
// ---------------------------------------------------------------------------
__global__ __launch_bounds__(256)
void mask_cast_kernel(const float* __restrict__ x, const int* __restrict__ mask,
                      __bf16* __restrict__ xb) {
    size_t i4   = (size_t)blockIdx.x * blockDim.x + threadIdx.x;
    size_t base = i4 * 4;
    int    row  = (int)(base >> 10);               // b*S + s  (D == 1024)
    float  m    = (float)mask[row];
    float4 v    = *(const float4*)(x + base);
    xb[base + 0] = (__bf16)(v.x * m);
    xb[base + 1] = (__bf16)(v.y * m);
    xb[base + 2] = (__bf16)(v.z * m);
    xb[base + 3] = (__bf16)(v.w * m);
}

// Kernel 2: weight fp32 -> bf16
__global__ __launch_bounds__(256)
void wcast_kernel(const float* __restrict__ w, __bf16* __restrict__ wb) {
    size_t base = ((size_t)blockIdx.x * blockDim.x + threadIdx.x) * 4;
    float4 v = *(const float4*)(w + base);
    wb[base + 0] = (__bf16)v.x;
    wb[base + 1] = (__bf16)v.y;
    wb[base + 2] = (__bf16)v.z;
    wb[base + 3] = (__bf16)v.w;
}

// ---------------------------------------------------------------------------
// Kernel 3: double-buffered bf16 WMMA GEMM  C[M,N] = A[M,K] * B[K,N]
//   block tile 128x64, BK=32, 8 waves; wave w owns rows [16w,16w+16) x 4 Ntiles
//   B tile staged COLUMN-major in LDS so B fragments are 2x ds_load_b128.
//   Ping-pong LDS: next tile's global loads issue before current-tile compute,
//   staged registers stored to the other buffer after the WMMAs, one barrier
//   per iteration -> HBM/L2 latency hidden behind matrix math.
//   mode 0: store bf16 at [b,h,s,u]   (Q / K head-major)
//   mode 1: store bf16 at [b,h,u,s]   (V transposed for PV B-operand)
//   mode 2: store fp32 row-major      (final output projection)
// ---------------------------------------------------------------------------
#define BM 128
#define BN 64
#define BK 32

__global__ __launch_bounds__(256)
void gemm_bf16_wmma(const __bf16* __restrict__ A, const __bf16* __restrict__ Bw,
                    void* __restrict__ Cout, int M, int N, int Kd, int mode) {
    __shared__ __bf16 As[2][BM * BK];   // 2 x 8 KB, row-major [m][k]
    __shared__ __bf16 Bs[2][BN * BK];   // 2 x 4 KB, col-major [n][k]

    const int tid  = threadIdx.x;
    const int wv   = tid >> 5;
    const int lane = tid & 31;
    const int khalf = lane >> 4;
    const int nl    = lane & 15;
    const int row0 = blockIdx.y * BM;
    const int col0 = blockIdx.x * BN;

    // staging coordinates (fixed per thread)
    const int ar0 = (tid) >> 2,        ao0 = (tid & 3) * 8;        // A chunk, pass 0
    const int ar1 = (256 + tid) >> 2,  ao1 = ((256 + tid) & 3) * 8;// A chunk, pass 1
    const int bk  = 2 * (tid >> 4);                                // B row pair
    const int bn0 = (tid & 15) * 4;                                // B col group

    bf16x8 areg0, areg1;
    bf16x4 breg0, breg1;

    auto gload = [&](int kk) {
        areg0 = *(const bf16x8*)&A[(size_t)(row0 + ar0) * Kd + kk + ao0];
        areg1 = *(const bf16x8*)&A[(size_t)(row0 + ar1) * Kd + kk + ao1];
        breg0 = *(const bf16x4*)&Bw[(size_t)(kk + bk) * N + col0 + bn0];
        breg1 = *(const bf16x4*)&Bw[(size_t)(kk + bk + 1) * N + col0 + bn0];
    };
    auto lstore = [&](int buf) {
        *(bf16x8*)&As[buf][ar0 * BK + ao0] = areg0;
        *(bf16x8*)&As[buf][ar1 * BK + ao1] = areg1;
        #pragma unroll
        for (int j = 0; j < 4; ++j) {     // pack (k,k+1) pairs -> col-major b32
            pk2 p2; p2.b[0] = breg0.v[j]; p2.b[1] = breg1.v[j];
            *(unsigned int*)&Bs[buf][(bn0 + j) * BK + bk] = p2.u;
        }
    };

    v8f acc[4] = {v8f{}, v8f{}, v8f{}, v8f{}};

    gload(0);
    lstore(0);
    __syncthreads();

    const int nIter = Kd / BK;
    for (int it = 0; it < nIter; ++it) {
        const int cur = it & 1;
        if (it + 1 < nIter) gload((it + 1) * BK);   // in-flight during compute

        // current tile: A fragment + all 4 B fragments first, then 4 WMMAs
        v16bf af = load_afrag(&As[cur][((wv << 4) + nl) * BK], khalf);
        v16bf bfr[4];
        #pragma unroll
        for (int ns = 0; ns < 4; ++ns)
            bfr[ns] = load_bfrag(&Bs[cur][(ns * 16 + nl) * BK + 16 * khalf]);
        #pragma unroll
        for (int ns = 0; ns < 4; ++ns)
            acc[ns] = __builtin_amdgcn_wmma_f32_16x16x32_bf16(
                false, af, false, bfr[ns], (short)0, acc[ns], false, false);

        if (it + 1 < nIter) lstore(cur ^ 1);        // after compute's DS reads
        __syncthreads();
    }

    // Writeout (C/D layout: element r -> row r + 8*khalf, col = lane&15)
    #pragma unroll
    for (int ns = 0; ns < 4; ++ns) {
        #pragma unroll
        for (int r = 0; r < 8; ++r) {
            const int gr = row0 + (wv << 4) + r + 8 * khalf;
            const int gc = col0 + ns * 16 + nl;
            const float val = acc[ns][r];
            if (mode == 2) {
                ((float*)Cout)[(size_t)gr * N + gc] = val;
            } else {
                const int b = gr >> 10, s = gr & 1023;    // S == 1024
                const int h = gc >> 6,  u = gc & 63;      // U == 64
                __bf16* O = (__bf16*)Cout;
                const size_t idx = (mode == 0)
                    ? ((((size_t)b * NB_H + h) * NB_S + s) * NB_U + u)
                    : ((((size_t)b * NB_H + h) * NB_U + u) * NB_S + s);
                O[idx] = (__bf16)val;
            }
        }
    }
}

// ---------------------------------------------------------------------------
// Kernel 4: flash attention, one 16-query tile per wave, 32-key chunks.
//   Q [B,H,S,U] bf16   (A operand: u-contiguous, two 16B runs per lane)
//   K [B,H,S,U] bf16   (QK^T B operand: fragment = 32 contiguous bytes of u)
//   Vt [B,H,U,S] bf16  (PV  B operand: fragment = 32 contiguous bytes of key)
//   Att [B,S,H*U] bf16
// ---------------------------------------------------------------------------
__global__ __launch_bounds__(256)
void attn_wmma(const __bf16* __restrict__ Q, const __bf16* __restrict__ K,
               const __bf16* __restrict__ Vt, const int* __restrict__ mask,
               __bf16* __restrict__ Att) {
    __shared__ __bf16 pls[8][16 * 32];   // per-wave P staging (C-layout -> A-layout)

    const int tid  = threadIdx.x;
    const int wv   = tid >> 5;
    const int lane = tid & 31;
    const int khalf = lane >> 4;
    const int nl    = lane & 15;

    const int wid = blockIdx.x * 8 + wv;      // global wave id
    const int b  = wid >> 10;                 // H * (S/16) = 1024 tiles / batch
    const int h  = (wid >> 6) & 15;
    const int qt = wid & 63;
    const size_t bh = (size_t)b * NB_H + h;

    // Q fragments (16 queries x 64 u = two 16x32 A-operands), loaded once
    const __bf16* Qp = Q + (bh * NB_S + (size_t)qt * 16 + nl) * NB_U;
    v16bf qf[2];
    qf[0] = load_afrag(Qp, khalf);
    qf[1] = load_afrag(Qp + 32, khalf);

    float mrow[8], lrow[8];
    #pragma unroll
    for (int r = 0; r < 8; ++r) { mrow[r] = -1e30f; lrow[r] = 0.f; }
    v8f o[4] = {v8f{}, v8f{}, v8f{}, v8f{}};

    const float scale = 0.125f;               // 1/sqrt(64)

    for (int kc = 0; kc < NB_S / 32; ++kc) {
        // prefetch next chunks of K and Vt into caches (global_prefetch_b8)
        if (kc + 1 < NB_S / 32) {
            __builtin_prefetch(&K[(bh * NB_S + (size_t)(kc + 1) * 32) * NB_U], 0, 0);
            __builtin_prefetch(&Vt[bh * NB_U * NB_S + (size_t)(kc + 1) * 32], 0, 0);
        }

        // ---- logits: all 4 K fragments first, then 4 chained WMMAs ----
        v16bf kfr[2][2];
        #pragma unroll
        for (int t = 0; t < 2; ++t) {
            const int key = kc * 32 + t * 16 + nl;
            const __bf16* Kp = K + (bh * NB_S + key) * NB_U + 16 * khalf;
            kfr[t][0] = load_bfrag(Kp);
            kfr[t][1] = load_bfrag(Kp + 32);
        }
        v8f s[2] = {v8f{}, v8f{}};
        #pragma unroll
        for (int t = 0; t < 2; ++t)
            #pragma unroll
            for (int c = 0; c < 2; ++c)
                s[t] = __builtin_amdgcn_wmma_f32_16x16x32_bf16(
                    false, qf[c], false, kfr[t][c], (short)0, s[t], false, false);

        // ---- hoist V fragment loads: in flight during softmax VALU work ----
        v16bf vfr[4];
        #pragma unroll
        for (int us = 0; us < 4; ++us)
            vfr[us] = load_bfrag(Vt + (bh * NB_U + us * 16 + nl) * NB_S +
                                 kc * 32 + 16 * khalf);

        // ---- masked online softmax ----
        const int   m0 = mask[b * NB_S + kc * 32 + nl];
        const int   m1 = mask[b * NB_S + kc * 32 + 16 + nl];
        const float bias0 = m0 ? 0.f : -1e30f, flag0 = m0 ? 1.f : 0.f;
        const float bias1 = m1 ? 0.f : -1e30f, flag1 = m1 ? 1.f : 0.f;

        float alpha[8];
        #pragma unroll
        for (int r = 0; r < 8; ++r) {
            const float a0 = s[0][r] * scale + bias0;
            const float a1 = s[1][r] * scale + bias1;
            s[0][r] = a0; s[1][r] = a1;
            const float cm = redmax16(fmaxf(a0, a1));
            const float nm = fmaxf(mrow[r], cm);
            alpha[r] = __expf(mrow[r] - nm);
            mrow[r]  = nm;
        }
        #pragma unroll
        for (int r = 0; r < 8; ++r) {
            const float p0 = flag0 * __expf(s[0][r] - mrow[r]);
            const float p1 = flag1 * __expf(s[1][r] - mrow[r]);
            const int prow = r + 8 * khalf;
            pls[wv][prow * 32 + nl]      = (__bf16)p0;   // wave-private LDS,
            pls[wv][prow * 32 + 16 + nl] = (__bf16)p1;   // DS in-order per wave
            lrow[r] = lrow[r] * alpha[r] + redsum16(p0 + p1);
        }
        #pragma unroll
        for (int us = 0; us < 4; ++us)
            #pragma unroll
            for (int r = 0; r < 8; ++r) o[us][r] *= alpha[r];

        // ---- P (16x32) A-operand from LDS (2x ds_load_b128), then PV ----
        v16bf pf = load_afrag(&pls[wv][nl * 32], khalf);

        #pragma unroll
        for (int us = 0; us < 4; ++us)
            o[us] = __builtin_amdgcn_wmma_f32_16x16x32_bf16(
                false, pf, false, vfr[us], (short)0, o[us], false, false);
    }

    // ---- normalize (softmax denom + post-mask renorm, eps guard) & store ----
    #pragma unroll
    for (int r = 0; r < 8; ++r) {
        const float inv = 1.f / (lrow[r] + 1e-7f);
        const int q = qt * 16 + r + 8 * khalf;
        #pragma unroll
        for (int us = 0; us < 4; ++us) {
            const int u = h * NB_U + us * 16 + nl;
            Att[((size_t)b * NB_S + q) * (NB_H * NB_U) + u] = (__bf16)(o[us][r] * inv);
        }
    }
}

// ---------------------------------------------------------------------------
// Launch
// ---------------------------------------------------------------------------
extern "C" void kernel_launch(void* const* d_in, const int* in_sizes, int n_in,
                              void* d_out, int out_size, void* d_ws, size_t ws_size,
                              hipStream_t stream) {
    const float* inputs = (const float*)d_in[0];
    const int*   mask   = (const int*)d_in[1];
    const float* Wq     = (const float*)d_in[2];
    const float* Wk     = (const float*)d_in[3];
    const float* Wv     = (const float*)d_in[4];
    const float* Wo     = (const float*)d_in[5];
    float*       out    = (float*)d_out;

    char* ws = (char*)d_ws;
    __bf16* xb  = (__bf16*)ws;                           // 16 MiB  [B*S, D]
    __bf16* wqb = (__bf16*)(ws + (16u << 20));           //  2 MiB
    __bf16* wkb = wqb + (1u << 20);
    __bf16* wvb = wkb + (1u << 20);
    __bf16* wob = wvb + (1u << 20);
    __bf16* Qb  = (__bf16*)(ws + (24u << 20));           // 16 MiB  [B,H,S,U]
    __bf16* Kb  = (__bf16*)(ws + (40u << 20));           // 16 MiB  [B,H,S,U]
    __bf16* Vtb = (__bf16*)(ws + (56u << 20));           // 16 MiB  [B,H,U,S]
    __bf16* Att = (__bf16*)(ws + (72u << 20));           // 16 MiB  [B*S, H*U]

    const int M = NB_B * NB_S;        // 8192
    const int N = NB_H * NB_U;        // 1024
    const int K = NB_D;               // 1024

    // 1. input mask + bf16 cast
    mask_cast_kernel<<<(size_t)NB_B * NB_S * NB_D / 4 / 256, 256, 0, stream>>>(
        inputs, mask, xb);

    // 2. weight casts
    const int wblocks = (NB_D * N) / 4 / 256;   // 1024
    wcast_kernel<<<wblocks, 256, 0, stream>>>(Wq, wqb);
    wcast_kernel<<<wblocks, 256, 0, stream>>>(Wk, wkb);
    wcast_kernel<<<wblocks, 256, 0, stream>>>(Wv, wvb);
    wcast_kernel<<<wblocks, 256, 0, stream>>>(Wo, wob);

    // 3. QKV projections (WMMA GEMM)
    dim3 ggrid(N / BN, M / BM);       // (16, 64)
    gemm_bf16_wmma<<<ggrid, 256, 0, stream>>>(xb, wqb, Qb,  M, N, K, 0);
    gemm_bf16_wmma<<<ggrid, 256, 0, stream>>>(xb, wkb, Kb,  M, N, K, 0);
    gemm_bf16_wmma<<<ggrid, 256, 0, stream>>>(xb, wvb, Vtb, M, N, K, 1);

    // 4. flash attention (1024 blocks x 8 waves = one 16-query tile per wave)
    attn_wmma<<<NB_B * NB_H * (NB_S / 16) / 8, 256, 0, stream>>>(
        Qb, Kb, Vtb, mask, Att);

    // 5. output projection -> fp32
    gemm_bf16_wmma<<<ggrid, 256, 0, stream>>>(Att, wob, out, M, N, K, 2);
}
